// DeepIRTModel_14834817040435
// MI455X (gfx1250) — compile-verified, matrix operand order checked by reading, not run on hardware
//
#include <hip/hip_runtime.h>
#include <cmath>

// ---------------------------------------------------------------------------
// DKVMN forward for MI455X (gfx1250, wave32, WMMA + TDM).
// 128 persistent workgroups x 256 threads; each WG owns BT=16 batch rows and
// scans all S=200 timesteps. Mv state (80MB, L2-resident) lives in d_ws.
// Gate/MLP weights are staged once per WG into dynamic LDS with
// TENSOR_LOAD_TO_LDS (s_wait_tensorcnt), GEMMs use v_wmma_f32_16x16x32_f16,
// and the bandwidth-critical Mv read/update pass is float4-vectorized.
// ---------------------------------------------------------------------------

#define B_    2048
#define S_    200
#define MEM_  50
#define KDIM_ 50
#define VDIM_ 200
#define FC_   50

#define BT      16
#define NWG     (B_ / BT)     // 128 workgroups
#define THREADS 256           // 8 waves of 32

// padded tile dims (K pads multiple of 32, N pads multiple of 16)
#define KQ_PAD   64           // KDIM 50  -> 64   (attention K)
#define KQA_PAD  224          // VDIM 200 -> 224  (erase/add K)
#define NE_PAD   208          // VDIM 200 -> 208  (erase/add N: 13 tiles)
#define KCAT_PAD 256          // 250      -> 256  (pred1 K)
#define NFC_PAD  64           // FC 50    -> 64   (pred1 N: 4 tiles)

#define EW_ELEMS   (NE_PAD * KQA_PAD)     // 46592 f16 per gate weight
#define PW_ELEMS   (NFC_PAD * KCAT_PAD)   // 16384 f16
#define DYN_ELEMS  (2 * EW_ELEMS + PW_ELEMS)
#define DYN_BYTES  (DYN_ELEMS * 2)        // 219136 B dynamic LDS

typedef _Float16 v16h __attribute__((ext_vector_type(16)));
typedef float    v8f  __attribute__((ext_vector_type(8)));

// TDM availability (device pass only; host pass parses the fallback branch).
#if defined(__gfx1250__) && __has_builtin(__builtin_amdgcn_tensor_load_to_lds) && \
    __has_builtin(__builtin_amdgcn_s_wait_tensorcnt) && \
    __has_builtin(__builtin_amdgcn_groupstaticsize)
#define USE_TDM 1
#else
#define USE_TDM 0
#endif

#if USE_TDM
typedef unsigned int tdm_u32x4 __attribute__((ext_vector_type(4)));
typedef int          tdm_i32x8 __attribute__((ext_vector_type(8)));
typedef int          tdm_i32x4 __attribute__((ext_vector_type(4)));

// Issue one 2D TENSOR_LOAD_TO_LDS: contiguous HxW f16 tile (tile == tensor).
// D# layout per CDNA5 ISA ch.8: group0 = {count, lds_addr, global_addr, type},
// group1 = {data_size, tensor_dim0/1, tile_dim0/1, dim0_stride}.
__device__ __forceinline__ void tdm_load_2d(unsigned lds_off, const void* gptr,
                                            unsigned W, unsigned H) {
    unsigned long long ga = (unsigned long long)(uintptr_t)gptr;
    tdm_u32x4 g0;
    g0[0] = 1u;                                   // count=1, user descriptor
    g0[1] = lds_off;                              // lds_addr (bytes)
    g0[2] = (unsigned)(ga & 0xffffffffu);         // global_addr[31:0]
    g0[3] = (unsigned)((ga >> 32) & 0x01ffffffu)  // global_addr[56:32]
          | (2u << 30);                           // type=2 ("image")
    tdm_i32x8 g1;
    g1[0] = (int)(1u << 16);                      // wg_mask=0, data_size=1 (2B)
    g1[1] = (int)((W & 0xffffu) << 16);           // tensor_dim0[15:0]
    g1[2] = (int)(((W >> 16) & 0xffffu) | ((H & 0xffffu) << 16)); // td0 hi, td1 lo
    g1[3] = (int)(((H >> 16) & 0xffffu) | ((W & 0xffffu) << 16)); // td1 hi, tile_dim0
    g1[4] = (int)(H & 0xffffu);                   // tile_dim1=H, tile_dim2=0
    g1[5] = (int)W;                               // tensor_dim0_stride[31:0]
    g1[6] = 0;                                    // stride0 hi, stride1 lo
    g1[7] = 0;                                    // stride1 hi
    tdm_i32x4 gz = {0, 0, 0, 0};
#if __clang_major__ >= 23
    tdm_i32x8 gz8 = {0, 0, 0, 0, 0, 0, 0, 0};
    __builtin_amdgcn_tensor_load_to_lds(g0, g1, gz, gz, gz8, 0);
#else
    __builtin_amdgcn_tensor_load_to_lds(g0, g1, gz, gz, 0);
#endif
}
#endif // USE_TDM

// Load a 16x32 (rows x K) f16 fragment from a row-major [rows][ldk] buffer.
// Per-lane layout per ISA 7.12.2: row = lane%16,
// K pairs = {0,2,4,6,16,18,20,22} + 8*(lane/16), each VGPR holds (k,k+1).
__device__ __forceinline__ v16h load_frag16(const _Float16* base, int ldk, int k0) {
    const int lane = threadIdx.x & 31;
    const int row  = lane & 15;
    const _Float16* p = base + row * ldk + k0 + ((lane >> 4) << 3);
    v16h f;
#pragma unroll
    for (int g = 0; g < 4; ++g) { f[2*g]   = p[2*g];      f[2*g+1]   = p[2*g+1]; }
#pragma unroll
    for (int g = 0; g < 4; ++g) { f[8+2*g] = p[16+2*g];   f[8+2*g+1] = p[16+2*g+1]; }
    return f;
}

__device__ __forceinline__ float sigmoidf_(float x) { return 1.f / (1.f + __expf(-x)); }

// ---------------------------------------------------------------------------
// Prologue: transpose+convert weights to padded f16 (N-major), zero the three
// dummy output arrays. Runs every call (graph-capture safe, deterministic).
// ---------------------------------------------------------------------------
__global__ void dkvmn_prep(const float* __restrict__ erase_w,
                           const float* __restrict__ add_w,
                           const float* __restrict__ pred_w1,
                           _Float16* __restrict__ eraseT,
                           _Float16* __restrict__ addT,
                           _Float16* __restrict__ predw1T,
                           float* __restrict__ out_zero, int nzero) {
    const int i = blockIdx.x * blockDim.x + threadIdx.x;
    if (i < EW_ELEMS) {
        int n = i / KQA_PAD, k = i - n * KQA_PAD;
        bool ok = (n < VDIM_) && (k < VDIM_);
        eraseT[i] = (_Float16)(ok ? erase_w[k * VDIM_ + n] : 0.f);
        addT[i]   = (_Float16)(ok ? add_w  [k * VDIM_ + n] : 0.f);
    } else if (i < EW_ELEMS + PW_ELEMS) {
        int j = i - EW_ELEMS;
        int n = j / KCAT_PAD, k = j - n * KCAT_PAD;
        bool ok = (n < FC_) && (k < (VDIM_ + KDIM_));
        predw1T[j] = (_Float16)(ok ? pred_w1[k * FC_ + n] : 0.f);
    }
    if (i < nzero) out_zero[i] = 0.f;
}

// ---------------------------------------------------------------------------
// Main persistent scan kernel.
// ---------------------------------------------------------------------------
__global__ __launch_bounds__(THREADS)
void dkvmn_main(const int* __restrict__ q_data,
                const int* __restrict__ qa_data,
                const float* __restrict__ q_embed_w,
                const float* __restrict__ qa_embed_w,
                const float* __restrict__ key_memory,
                const float* __restrict__ init_value_memory,
                const float* __restrict__ erase_b,
                const float* __restrict__ add_b,
                const float* __restrict__ pred_b1,
                const float* __restrict__ pred_w2,
                const float* __restrict__ pred_b2,
                float* __restrict__ Mv,
                const _Float16* __restrict__ eraseT,
                const _Float16* __restrict__ addT,
                const _Float16* __restrict__ predw1T,
                float* __restrict__ out) {
    __shared__ _Float16 sQh [BT][KQ_PAD];    // q embedding, f16, zero-padded K
    __shared__ _Float16 sQAh[BT][KQA_PAD];   // qa embedding, f16, zero-padded K
    __shared__ float    sW  [BT][KQ_PAD];    // attention scores -> softmax w
    __shared__ float    sE  [BT][NE_PAD];    // erase gate
    __shared__ float    sA  [BT][NE_PAD];    // add gate
    __shared__ _Float16 sCat[BT][KCAT_PAD];  // [read | q] f16, zero-padded
    __shared__ float    sH  [BT][NFC_PAD];   // MLP hidden
    __shared__ _Float16 sKey[KQ_PAD][KQ_PAD];// key_memory [n][k] f16 padded
    extern __shared__ _Float16 dynlds[];     // weight cache: erase|add|pred1

    const int tid   = threadIdx.x;
    const int wave  = tid >> 5;
    const int lane  = tid & 31;
    const int bbase = blockIdx.x * BT;

    // Stage key matrix once (sKey[n][k] == B^T for scores = q @ K^T).
    for (int idx = tid; idx < KQ_PAD * KQ_PAD; idx += THREADS) {
        int n = idx >> 6, k = idx & 63;
        float v = (n < MEM_ && k < KDIM_) ? key_memory[n * KDIM_ + k] : 0.f;
        sKey[n][k] = (_Float16)v;
    }
    // Stage gate/MLP weights into dynamic LDS, once per WG.
#if USE_TDM
    if (wave == 0) {
        unsigned base = (__builtin_amdgcn_groupstaticsize() + 15u) & ~15u;
        tdm_load_2d(base,                       eraseT,  KQA_PAD,  NE_PAD);
        tdm_load_2d(base + 2u * EW_ELEMS,       addT,    KQA_PAD,  NE_PAD);
        tdm_load_2d(base + 4u * EW_ELEMS,       predw1T, KCAT_PAD, NFC_PAD);
        __builtin_amdgcn_s_wait_tensorcnt((short)0);
    }
#else
    for (int i = tid; i < EW_ELEMS; i += THREADS) dynlds[i]             = eraseT[i];
    for (int i = tid; i < EW_ELEMS; i += THREADS) dynlds[EW_ELEMS + i]  = addT[i];
    for (int i = tid; i < PW_ELEMS; i += THREADS) dynlds[2*EW_ELEMS + i] = predw1T[i];
#endif
    const _Float16* eraseL = dynlds;
    const _Float16* addL   = dynlds + EW_ELEMS;
    const _Float16* predL  = dynlds + 2 * EW_ELEMS;

    // Init this WG's Mv tile from init_value_memory (broadcast over batch).
    {
        float* mvb = Mv + (size_t)bbase * MEM_ * VDIM_;
        for (int i = tid; i < BT * MEM_ * VDIM_; i += THREADS)
            mvb[i] = init_value_memory[i % (MEM_ * VDIM_)];
    }
    __syncthreads();

    for (int t = 0; t < S_; ++t) {
        // -- P1: gather embeddings into LDS (f16, zero-padded) --------------
        for (int idx = tid; idx < BT * KQ_PAD; idx += THREADS) {
            int b = idx >> 6, k = idx & 63;
            int q = q_data[(bbase + b) * S_ + t];
            float v = (k < KDIM_) ? q_embed_w[q * KDIM_ + k] : 0.f;
            sQh[b][k] = (_Float16)v;
        }
        for (int idx = tid; idx < BT * KQA_PAD; idx += THREADS) {
            int b = idx / KQA_PAD, k = idx - b * KQA_PAD;
            int qa = qa_data[(bbase + b) * S_ + t];
            float v = (k < VDIM_) ? qa_embed_w[qa * VDIM_ + k] : 0.f;
            sQAh[b][k] = (_Float16)v;
        }
        if (t + 1 < S_ && tid < BT) {   // prefetch next step's embedding rows
            int qn  = q_data [(bbase + tid) * S_ + t + 1];
            int qan = qa_data[(bbase + tid) * S_ + t + 1];
            __builtin_prefetch(&q_embed_w[qn * KDIM_], 0, 1);
            __builtin_prefetch(&qa_embed_w[qan * VDIM_], 0, 1);
        }
        __syncthreads();

        // -- P2: attention scores [16,50] via WMMA (4 N-tiles, waves 0..3) --
        if (wave < 4) {
            const int n0 = wave * 16;
            v8f acc = {};
#pragma unroll
            for (int k0 = 0; k0 < KQ_PAD; k0 += 32) {
                v16h af = load_frag16(&sQh[0][0], KQ_PAD, k0);
                v16h bf = load_frag16(&sKey[n0][0], KQ_PAD, k0);
                acc = __builtin_amdgcn_wmma_f32_16x16x32_f16(
                          false, af, false, bf, (short)0, acc, false, false);
            }
            const int n  = lane & 15;
            const int mh = (lane >> 4) << 3;
#pragma unroll
            for (int g = 0; g < 8; ++g) sW[mh + g][n0 + n] = acc[g];
        }
        __syncthreads();

        // -- P3: softmax over MEM slots, one thread per batch row -----------
        if (tid < BT) {
            float mx = -1e30f;
            for (int m = 0; m < MEM_; ++m) mx = fmaxf(mx, sW[tid][m]);
            float s = 0.f;
            for (int m = 0; m < MEM_; ++m) {
                float e = __expf(sW[tid][m] - mx);
                sW[tid][m] = e; s += e;
            }
            float inv = 1.f / s;
            for (int m = 0; m < MEM_; ++m) sW[tid][m] *= inv;
            for (int m = MEM_; m < KQ_PAD; ++m) sW[tid][m] = 0.f;
        }
        __syncthreads();

        // -- P4: erase/add GEMMs [16,200]@[200,200] via WMMA (26 tile-jobs) -
        for (int job = wave; job < 26; job += 8) {
            const bool is_e = (job < 13);
            const _Float16* WT  = is_e ? eraseL  : addL;
            const float*    bia = is_e ? erase_b : add_b;
            float*          dst = is_e ? &sE[0][0] : &sA[0][0];
            const int n0 = (job % 13) * 16;
            v8f acc = {};
            for (int k0 = 0; k0 < KQA_PAD; k0 += 32) {
                v16h af = load_frag16(&sQAh[0][0], KQA_PAD, k0);
                v16h bf = load_frag16(WT + (size_t)n0 * KQA_PAD, KQA_PAD, k0);
                acc = __builtin_amdgcn_wmma_f32_16x16x32_f16(
                          false, af, false, bf, (short)0, acc, false, false);
            }
            const int n  = n0 + (lane & 15);
            const int mh = (lane >> 4) << 3;
            if (n < VDIM_) {
                float bv = bia[n];
#pragma unroll
                for (int g = 0; g < 8; ++g) {
                    float x = acc[g] + bv;
                    dst[(mh + g) * NE_PAD + n] = is_e ? sigmoidf_(x) : tanhf(x);
                }
            }
        }
        __syncthreads();

        // -- P5: fused batched read + Mv erase/add update, float4-wide ------
        {
            float* mvb = Mv + (size_t)bbase * MEM_ * VDIM_;
            for (int idx = tid; idx < BT * (VDIM_ / 4); idx += THREADS) {
                int b = idx / (VDIM_ / 4);
                int v = (idx - b * (VDIM_ / 4)) * 4;
                float4 ev = *(const float4*)&sE[b][v];
                float4 av = *(const float4*)&sA[b][v];
                float4 acc = make_float4(0.f, 0.f, 0.f, 0.f);
                float* mp = mvb + (size_t)b * MEM_ * VDIM_ + v;
#pragma unroll 2
                for (int m = 0; m < MEM_; ++m) {
                    float w = sW[b][m];
                    float4 mv4 = *(float4*)(mp + m * VDIM_);
                    acc.x += w * mv4.x;  acc.y += w * mv4.y;
                    acc.z += w * mv4.z;  acc.w += w * mv4.w;
                    mv4.x = mv4.x * (1.f - w * ev.x) + w * av.x;
                    mv4.y = mv4.y * (1.f - w * ev.y) + w * av.y;
                    mv4.z = mv4.z * (1.f - w * ev.z) + w * av.z;
                    mv4.w = mv4.w * (1.f - w * ev.w) + w * av.w;
                    *(float4*)(mp + m * VDIM_) = mv4;
                }
                sCat[b][v+0] = (_Float16)acc.x;
                sCat[b][v+1] = (_Float16)acc.y;
                sCat[b][v+2] = (_Float16)acc.z;
                sCat[b][v+3] = (_Float16)acc.w;
            }
            for (int idx = tid; idx < BT * (KCAT_PAD - VDIM_); idx += THREADS) {
                int b = idx / (KCAT_PAD - VDIM_), j = idx - b * (KCAT_PAD - VDIM_);
                sCat[b][VDIM_ + j] = (j < KDIM_) ? sQh[b][j] : (_Float16)0.f;
            }
        }
        __syncthreads();

        // -- P6: pred MLP layer 1 [16,250]@[250,50] via WMMA ----------------
        if (wave < 4) {
            const int n0 = wave * 16;
            v8f acc = {};
            for (int k0 = 0; k0 < KCAT_PAD; k0 += 32) {
                v16h af = load_frag16(&sCat[0][0], KCAT_PAD, k0);
                v16h bf = load_frag16(predL + (size_t)n0 * KCAT_PAD, KCAT_PAD, k0);
                acc = __builtin_amdgcn_wmma_f32_16x16x32_f16(
                          false, af, false, bf, (short)0, acc, false, false);
            }
            const int n  = n0 + (lane & 15);
            const int mh = (lane >> 4) << 3;
#pragma unroll
            for (int g = 0; g < 8; ++g)
                sH[mh + g][n] = (n < FC_) ? tanhf(acc[g] + pred_b1[n]) : 0.f;
        }
        __syncthreads();

        // -- P7: pred head + write output -----------------------------------
        if (tid < BT) {
            float s = pred_b2[0];
            for (int j = 0; j < FC_; ++j) s += sH[tid][j] * pred_w2[j];
            out[(bbase + tid) * S_ + t] = sigmoidf_(s);
        }
        __syncthreads();   // protect LDS before next timestep overwrites it
    }
}

// ---------------------------------------------------------------------------
extern "C" void kernel_launch(void* const* d_in, const int* in_sizes, int n_in,
                              void* d_out, int out_size, void* d_ws, size_t ws_size,
                              hipStream_t stream) {
    const int*   q_data    = (const int*)  d_in[0];
    const int*   qa_data   = (const int*)  d_in[1];
    const float* q_embed_w = (const float*)d_in[2];
    const float* qa_embed_w= (const float*)d_in[3];
    const float* key_mem   = (const float*)d_in[4];
    const float* init_vm   = (const float*)d_in[5];
    const float* erase_w   = (const float*)d_in[6];
    const float* erase_b   = (const float*)d_in[7];
    const float* add_w     = (const float*)d_in[8];
    const float* add_b     = (const float*)d_in[9];
    const float* pred_w1   = (const float*)d_in[10];
    const float* pred_b1   = (const float*)d_in[11];
    const float* pred_w2   = (const float*)d_in[12];
    const float* pred_b2   = (const float*)d_in[13];

    float* out = (float*)d_out;

    // Workspace layout: Mv state (80 MB) + f16 transposed weights.
    char*  ws   = (char*)d_ws;
    float* Mv   = (float*)ws;
    size_t off  = (size_t)B_ * MEM_ * VDIM_ * sizeof(float);
    _Float16* eraseT  = (_Float16*)(ws + off); off += (size_t)EW_ELEMS * 2;
    _Float16* addT    = (_Float16*)(ws + off); off += (size_t)EW_ELEMS * 2;
    _Float16* predw1T = (_Float16*)(ws + off);

    const int nzero = 3 * B_ * S_;                  // outputs 1..3 are zeros
    dkvmn_prep<<<(nzero + 255) / 256, 256, 0, stream>>>(
        erase_w, add_w, pred_w1, eraseT, addT, predw1T, out + (size_t)B_ * S_, nzero);

    (void)hipFuncSetAttribute(reinterpret_cast<const void*>(&dkvmn_main),
                              hipFuncAttributeMaxDynamicSharedMemorySize, DYN_BYTES);
    dkvmn_main<<<NWG, THREADS, DYN_BYTES, stream>>>(
        q_data, qa_data, q_embed_w, qa_embed_w, key_mem, init_vm,
        erase_b, add_b, pred_b1, pred_w2, pred_b2,
        Mv, eraseT, addT, predw1T, out);
}
